// URATMIL_4440996184099
// MI455X (gfx1250) — compile-verified, matrix operand order checked by reading, not run on hardware
//
#include <hip/hip_runtime.h>
#include <hip/hip_bf16.h>

// ---------------- problem constants ----------------
constexpr int kN    = 100000;
constexpr int kFEAT = 512;
constexpr int kD    = 256;
constexpr int kA    = 128;
constexpr int kH    = 4;
constexpr int kP    = 64;
constexpr int kD2   = 512;          // 2*D
constexpr float kALPHA      = 0.5f;
constexpr float kDELTA_ART  = 0.6f;
constexpr float kDELTA_NEAR = 0.4f;

// output layout (floats): logits(2) phat(2) z_bag(256) attn_w(100000) kl(1) recon(1)
constexpr int OUT_LOGITS = 0;
constexpr int OUT_PHAT   = 2;
constexpr int OUT_ZBAG   = 4;
constexpr int OUT_ATTN   = 4 + kD;
constexpr int OUT_KL     = OUT_ATTN + kN;
constexpr int OUT_RL     = OUT_KL + 1;

// ---------------- WMMA types ----------------
typedef __attribute__((ext_vector_type(16))) __bf16 v16bf;
typedef __attribute__((ext_vector_type(8)))  float  v8f;

// ---------------- small helpers ----------------
__device__ __forceinline__ unsigned short f2bf_bits(float f) {
  __bf16 b = (__bf16)f;                      // hardware f32->bf16 convert
  unsigned short u; __builtin_memcpy(&u, &b, 2); return u;
}
__device__ __forceinline__ float bf2f(unsigned short h) {
  union { unsigned u; float f; } v; v.u = ((unsigned)h) << 16;
  return v.f;
}
__device__ __forceinline__ float gelu_exact(float x) {
  return 0.5f * x * (1.0f + erff(x * 0.70710678118654752f));
}
// monotonic float<->uint for atomicMax on floats (incl. negatives)
__device__ __forceinline__ unsigned ford(float f) {
  union { float f; unsigned u; } v; v.f = f;
  return (v.u & 0x80000000u) ? ~v.u : (v.u | 0x80000000u);
}
__device__ __forceinline__ float funord(unsigned u) {
  union { unsigned u; float f; } v;
  v.u = (u & 0x80000000u) ? (u & 0x7FFFFFFFu) : ~u;
  return v.f;
}

// Swizzled fragment storage: per 16x32 (A) or 32x16 (B) tile, 32 lanes x 16 bf16
// contiguous (32B per lane). One fragment load = two b128 loads.
__device__ __forceinline__ v16bf ld_frag(const unsigned short* p) {
  union { uint4 q[2]; v16bf v; } u;
  u.q[0] = *((const uint4*)p);
  u.q[1] = *((const uint4*)p + 1);
  return u.v;
}
// element (m, col) -> index inside a swizzled row-tile (col across full K width)
__device__ __forceinline__ int swz_idx(int m, int col) {
  int kt = col >> 5, kl = col & 31;
  int lane = m + ((kl & 8) << 1);            // +16 if kl&8
  int i = (kl & 7) | ((kl & 16) >> 1);
  return (kt * 32 + lane) * 16 + i;
}

#define WMMA_BF16(acc, a, b) \
  __builtin_amdgcn_wmma_f32_16x16x32_bf16(false, (a), false, (b), (short)0, (acc), false, false)

// ================= K0: init accumulators + attn_w region =================
__global__ void k_init(float* attn_w, float* accf, unsigned* hmax, int accf_cnt) {
  int i = blockIdx.x * blockDim.x + threadIdx.x;
  if (i < kN) attn_w[i] = 0.0f;
  if (i < accf_cnt) accf[i] = 0.0f;
  if (i < kH) hmax[i] = 0u;
}

// ================= prototypes: normalize to linear bf16 =================
__global__ void k_pn(const float* __restrict__ protos, unsigned short* __restrict__ pn) {
  int p = blockIdx.x;           // 0..P-1
  int d = threadIdx.x;          // 0..255
  __shared__ float s_sq;
  if (d == 0) s_sq = 0.0f;
  __syncthreads();
  float v = protos[(size_t)p * kD + d];
  atomicAdd(&s_sq, v * v);
  __syncthreads();
  pn[(size_t)p * kD + d] = f2bf_bits(v * rsqrtf(s_sq + 1e-24f));
}

// ===== weight swizzler: f32 row-major (K x Ncols) -> bf16 B-fragment layout =====
// tile (kt, nt) stored at ((nt*KT + kt)*32 + lane)*16
__global__ void k_wswz(const float* __restrict__ src, unsigned short* __restrict__ dst,
                       int K, int Ncols) {
  int wv = blockIdx.x * (blockDim.x >> 5) + (threadIdx.x >> 5);
  int lane = threadIdx.x & 31;
  int KT = K >> 5, NT = Ncols >> 4;
  if (wv >= KT * NT) return;
  int nt = wv / KT, kt = wv % KT;
  const float* p = src + (size_t)(kt * 32) * Ncols + nt * 16;
  int n = lane & 15, kb = (lane >> 4) * 8;
  union { unsigned short s[16]; uint4 q[2]; } u;
#pragma unroll
  for (int i = 0; i < 16; ++i) {
    int k = kb + (i < 8 ? i : i + 8);
    u.s[i] = f2bf_bits(p[(size_t)k * Ncols + n]);
  }
  unsigned short* o = dst + ((size_t)wv * 32 + lane) * 16;
  *((uint4*)o) = u.q[0];
  *((uint4*)o + 1) = u.q[1];
}
// same but source is transposed bf16 (Ncols x K row-major): B[k][n] = srcT[n*K+k]
__global__ void k_wswzT(const unsigned short* __restrict__ srcT, unsigned short* __restrict__ dst,
                        int K, int Ncols) {
  int wv = blockIdx.x * (blockDim.x >> 5) + (threadIdx.x >> 5);
  int lane = threadIdx.x & 31;
  int KT = K >> 5, NT = Ncols >> 4;
  if (wv >= KT * NT) return;
  int nt = wv / KT, kt = wv % KT;
  const unsigned short* p = srcT + (size_t)(nt * 16) * K + kt * 32;
  int n = lane & 15, kb = (lane >> 4) * 8;
  union { unsigned short s[16]; uint4 q[2]; } u;
#pragma unroll
  for (int i = 0; i < 16; ++i) {
    int k = kb + (i < 8 ? i : i + 8);
    u.s[i] = p[(size_t)n * K + k];
  }
  unsigned short* o = dst + ((size_t)wv * 32 + lane) * 16;
  *((uint4*)o) = u.q[0];
  *((uint4*)o + 1) = u.q[1];
}

// ================= K1: h = GELU(LN(feats @ pW + pb)) -> swizzled bf16 =========
// grid = N/16 blocks, 128 threads (4 waves); wave covers 64 cols (4 col tiles)
__global__ void __launch_bounds__(128)
k1_proj(const float* __restrict__ feats, const unsigned short* __restrict__ pW_s,
        const float* __restrict__ pb, const float* __restrict__ png,
        const float* __restrict__ pnb, unsigned short* __restrict__ h_swz) {
  const int row0 = blockIdx.x * 16;
  const int tid  = threadIdx.x;
  const int lane = tid & 31;
  const int wave = tid >> 5;
  const int mlo  = 8 * (lane >> 4);
  const int KT   = kFEAT / 32;   // 16

  __shared__ __align__(16) unsigned short h_s[16 * kD];  // swizzled pre-LN tile
  __shared__ float s_sum[16], s_sq[16];
  if (tid < 16) { s_sum[tid] = 0.0f; s_sq[tid] = 0.0f; }
  __syncthreads();

  v8f acc[4] = {};
  for (int k0 = 0; k0 < kFEAT; k0 += 32) {
    // A fragment from fp32 feats: two runs of 8 consecutive floats per lane
    const float* r0 = feats + (size_t)(row0 + (lane & 15)) * kFEAT + k0 + mlo;
    float4 a0 = *((const float4*)r0);
    float4 a1 = *((const float4*)(r0 + 4));
    float4 a2 = *((const float4*)(r0 + 16));
    float4 a3 = *((const float4*)(r0 + 20));
    v16bf a;
    a[0] = (__bf16)a0.x;  a[1] = (__bf16)a0.y;  a[2] = (__bf16)a0.z;  a[3] = (__bf16)a0.w;
    a[4] = (__bf16)a1.x;  a[5] = (__bf16)a1.y;  a[6] = (__bf16)a1.z;  a[7] = (__bf16)a1.w;
    a[8] = (__bf16)a2.x;  a[9] = (__bf16)a2.y;  a[10] = (__bf16)a2.z; a[11] = (__bf16)a2.w;
    a[12] = (__bf16)a3.x; a[13] = (__bf16)a3.y; a[14] = (__bf16)a3.z; a[15] = (__bf16)a3.w;
    int kt = k0 >> 5;
#pragma unroll
    for (int t = 0; t < 4; ++t) {
      int nt = wave * 4 + t;
      v16bf b = ld_frag(pW_s + ((size_t)(nt * KT + kt) * 32 + lane) * 16);
      acc[t] = WMMA_BF16(acc[t], a, b);
    }
  }
  // bias + stage raw tile + per-row partial sums (pre-reduced across col tiles)
  float ps[8] = {0,0,0,0,0,0,0,0}, pq[8] = {0,0,0,0,0,0,0,0};
#pragma unroll
  for (int t = 0; t < 4; ++t) {
    int col = wave * 64 + t * 16 + (lane & 15);
    float bias = pb[col];
#pragma unroll
    for (int r = 0; r < 8; ++r) {
      float v = acc[t][r] + bias;
      h_s[swz_idx(mlo + r, col)] = f2bf_bits(v);
      ps[r] += v; pq[r] += v * v;
    }
  }
#pragma unroll
  for (int r = 0; r < 8; ++r) {
    atomicAdd(&s_sum[mlo + r], ps[r]);
    atomicAdd(&s_sq[mlo + r], pq[r]);
  }
  __syncthreads();
  // normalize + GELU, write coalesced swizzled groups (2 groups of 16 per thread)
#pragma unroll
  for (int c = 0; c < 2; ++c) {
    int g = tid * 2 + c;                 // 256 groups = 8 kt x 32 lanes
    int kt = g >> 5, lane_s = g & 31, m = lane_s & 15, kb = (lane_s >> 4) * 8;
    float mean = s_sum[m] * (1.0f / kD);
    float var  = s_sq[m] * (1.0f / kD) - mean * mean;
    float rstd = rsqrtf(var + 1e-5f);
    union { unsigned short s[16]; uint4 q[2]; } u;
    u.q[0] = *((const uint4*)(h_s + g * 16));
    u.q[1] = *((const uint4*)(h_s + g * 16) + 1);
#pragma unroll
    for (int i = 0; i < 16; ++i) {
      int col = kt * 32 + kb + (i < 8 ? i : i + 8);
      float x = (bf2f(u.s[i]) - mean) * rstd * png[col] + pnb[col];
      u.s[i] = f2bf_bits(gelu_exact(x));
    }
    unsigned short* o = h_swz + (size_t)blockIdx.x * (16 * kD) + g * 16;
    *((uint4*)o) = u.q[0];
    *((uint4*)o + 1) = u.q[1];
  }
}

// ================= K2: fused mu/lv/OOD/decoder/attention per 16-row tile ====
__global__ void __launch_bounds__(128)
k2_main(const unsigned short* __restrict__ h_swz,
        const unsigned short* __restrict__ muW_s, const float* __restrict__ mub,
        const unsigned short* __restrict__ lvW_s, const float* __restrict__ lvb,
        const unsigned short* __restrict__ pn_s,
        const unsigned short* __restrict__ dW1_s, const float* __restrict__ db1,
        const float* __restrict__ dng, const float* __restrict__ dnb,
        const unsigned short* __restrict__ dW2_s, const float* __restrict__ db2,
        const unsigned short* __restrict__ aW_s, const float* __restrict__ aWb,
        const float* __restrict__ aV, const float* __restrict__ aVb,
        unsigned short* __restrict__ z_g, float* __restrict__ u_g,
        float* __restrict__ mask_g, float* __restrict__ score_g,
        float* __restrict__ kl_g, float* __restrict__ rl_g, float* __restrict__ gate_g) {
  const int row0 = blockIdx.x * 16;
  const int tid  = threadIdx.x;
  const int lane = tid & 31;
  const int wave = tid >> 5;
  const int mlo  = 8 * (lane >> 4);
  const unsigned short* h_tile = h_swz + (size_t)blockIdx.x * (16 * kD);

  __shared__ __align__(16) unsigned short z_s[16 * kD];    // swizzled bf16 z tile
  __shared__ __align__(16) unsigned short t_s[16 * kD2];   // swizzled bf16 decoder tile
  __shared__ float s_zsq[16], s_usum[16], s_tsum[16], s_tsq[16], s_recon[16], s_ufin[16];
  __shared__ unsigned s_maxcos[16];
  __shared__ float s_score[kH * 16];
  __shared__ float s_kl, s_rl;

  if (tid < 16) {
    s_zsq[tid] = 0.0f; s_usum[tid] = 0.0f; s_tsum[tid] = 0.0f;
    s_tsq[tid] = 0.0f; s_recon[tid] = 0.0f; s_maxcos[tid] = 0u;
  }
  if (tid < kH * 16) s_score[tid] = 0.0f;
  if (tid == 0) { s_kl = 0.0f; s_rl = 0.0f; }
  __syncthreads();

  // ---- Stage A: z = h @ muW + mub ----
  {
    v8f acc[4] = {};
#pragma unroll
    for (int kt = 0; kt < 8; ++kt) {
      v16bf a = ld_frag(h_tile + (kt * 32 + lane) * 16);
#pragma unroll
      for (int t = 0; t < 4; ++t) {
        int nt = wave * 4 + t;
        v16bf b = ld_frag(muW_s + ((size_t)(nt * 8 + kt) * 32 + lane) * 16);
        acc[t] = WMMA_BF16(acc[t], a, b);
      }
    }
    float zp[8] = {0,0,0,0,0,0,0,0};
#pragma unroll
    for (int t = 0; t < 4; ++t) {
      int col = wave * 64 + t * 16 + (lane & 15);
      float bias = mub[col];
#pragma unroll
      for (int r = 0; r < 8; ++r) {
        int m = mlo + r;
        float zv = acc[t][r] + bias;
        unsigned short zb = f2bf_bits(zv);
        z_s[swz_idx(m, col)] = zb;
        z_g[(size_t)(row0 + m) * kD + col] = zb;
        zp[r] += zv * zv;
      }
    }
#pragma unroll
    for (int r = 0; r < 8; ++r) atomicAdd(&s_zsq[mlo + r], zp[r]);
  }
  __syncthreads();

  // ---- Stage B: log_var, u_total, KL partial ----
  {
    v8f acc[4] = {};
#pragma unroll
    for (int kt = 0; kt < 8; ++kt) {
      v16bf a = ld_frag(h_tile + (kt * 32 + lane) * 16);
#pragma unroll
      for (int t = 0; t < 4; ++t) {
        int nt = wave * 4 + t;
        v16bf b = ld_frag(lvW_s + ((size_t)(nt * 8 + kt) * 32 + lane) * 16);
        acc[t] = WMMA_BF16(acc[t], a, b);
      }
    }
    float klloc = 0.0f, up[8] = {0,0,0,0,0,0,0,0};
#pragma unroll
    for (int t = 0; t < 4; ++t) {
      int col = wave * 64 + t * 16 + (lane & 15);
      float bias = lvb[col];
#pragma unroll
      for (int r = 0; r < 8; ++r) {
        int m = mlo + r;
        float lv = fminf(4.0f, fmaxf(-8.0f, acc[t][r] + bias));
        float ev = __expf(lv);
        up[r] += ev;
        float mu = bf2f(z_s[swz_idx(m, col)]);
        klloc += 1.0f + lv - mu * mu - ev;
      }
    }
#pragma unroll
    for (int r = 0; r < 8; ++r) atomicAdd(&s_usum[mlo + r], up[r]);
    atomicAdd(&s_kl, klloc);
  }

  // ---- Stage C: cosine to prototypes (each wave: 16 protos = 1 col tile) ----
  {
    v8f pc = {};
#pragma unroll
    for (int kt = 0; kt < 8; ++kt) {
      v16bf a = ld_frag(z_s + (kt * 32 + lane) * 16);
      v16bf b = ld_frag(pn_s + ((size_t)(wave * 8 + kt) * 32 + lane) * 16);
      pc = WMMA_BF16(pc, a, b);
    }
#pragma unroll
    for (int r = 0; r < 8; ++r) {
      int m = mlo + r;
      float cosv = pc[r] * rsqrtf(s_zsq[m] + 1e-24f);
      atomicMax(&s_maxcos[m], ford(cosv));
    }
  }
  __syncthreads();

  // ---- Stage D: t = LN(ReLU(z @ dW1 + db1)) (each wave: 128 cols) ----
  {
    v8f td[8] = {};
#pragma unroll
    for (int kt = 0; kt < 8; ++kt) {
      v16bf a = ld_frag(z_s + (kt * 32 + lane) * 16);
#pragma unroll
      for (int t = 0; t < 8; ++t) {
        int nt = wave * 8 + t;
        v16bf b = ld_frag(dW1_s + ((size_t)(nt * 8 + kt) * 32 + lane) * 16);
        td[t] = WMMA_BF16(td[t], a, b);
      }
    }
    float ps[8] = {0,0,0,0,0,0,0,0}, pq[8] = {0,0,0,0,0,0,0,0};
#pragma unroll
    for (int t = 0; t < 8; ++t) {
      int col = wave * 128 + t * 16 + (lane & 15);
      float bias = db1[col];
#pragma unroll
      for (int r = 0; r < 8; ++r) {
        float tv = fmaxf(td[t][r] + bias, 0.0f);
        t_s[swz_idx(mlo + r, col)] = f2bf_bits(tv);
        ps[r] += tv; pq[r] += tv * tv;
      }
    }
#pragma unroll
    for (int r = 0; r < 8; ++r) {
      atomicAdd(&s_tsum[mlo + r], ps[r]);
      atomicAdd(&s_tsq[mlo + r], pq[r]);
    }
  }
  __syncthreads();
  // normalize t in place: 512 swizzled groups, 4 per thread (contiguous b128)
  {
#pragma unroll
    for (int c = 0; c < 4; ++c) {
      int g = tid * 4 + c;
      int kt = g >> 5, lane_s = g & 31, m = lane_s & 15, kb = (lane_s >> 4) * 8;
      float mean = s_tsum[m] * (1.0f / kD2);
      float var  = s_tsq[m] * (1.0f / kD2) - mean * mean;
      float rstd = rsqrtf(var + 1e-5f);
      union { unsigned short s[16]; uint4 q[2]; } u;
      u.q[0] = *((const uint4*)(t_s + g * 16));
      u.q[1] = *((const uint4*)(t_s + g * 16) + 1);
#pragma unroll
      for (int i = 0; i < 16; ++i) {
        int col = kt * 32 + kb + (i < 8 ? i : i + 8);
        float x = (bf2f(u.s[i]) - mean) * rstd * dng[col] + dnb[col];
        u.s[i] = f2bf_bits(x);
      }
      *((uint4*)(t_s + g * 16)) = u.q[0];
      *((uint4*)(t_s + g * 16) + 1) = u.q[1];
    }
  }
  __syncthreads();

  // ---- Stage E: h_hat = t @ dW2 + db2 ; recon err ----
  {
    v8f hh[4] = {};
#pragma unroll
    for (int kt = 0; kt < 16; ++kt) {
      v16bf a = ld_frag(t_s + (kt * 32 + lane) * 16);
#pragma unroll
      for (int t = 0; t < 4; ++t) {
        int nt = wave * 4 + t;
        v16bf b = ld_frag(dW2_s + ((size_t)(nt * 16 + kt) * 32 + lane) * 16);
        hh[t] = WMMA_BF16(hh[t], a, b);
      }
    }
    float rloc = 0.0f, rp[8] = {0,0,0,0,0,0,0,0};
#pragma unroll
    for (int t = 0; t < 4; ++t) {
      int col = wave * 64 + t * 16 + (lane & 15);
      float bias = db2[col];
#pragma unroll
      for (int r = 0; r < 8; ++r) {
        int m = mlo + r;
        float v = hh[t][r] + bias;
        float hv = bf2f(h_tile[swz_idx(m, col)]);
        float d = hv - v;
        rp[r] += d * d; rloc += d * d;
      }
    }
#pragma unroll
    for (int r = 0; r < 8; ++r) atomicAdd(&s_recon[mlo + r], rp[r]);
    atomicAdd(&s_rl, rloc);
  }
  __syncthreads();

  // ---- Stage F: per-row u_total, gate, global partials ----
  if (tid < 16) {
    int m = tid;
    float u = s_usum[m] * (1.0f / kD);
    u_g[row0 + m] = u;
    s_ufin[m] = fmaxf(u, 1e-6f);
    float sot  = 1.0f - funord(s_maxcos[m]);
    float rerr = s_recon[m] * (1.0f / kD);
    float sood = (0.5f + kALPHA) * rerr;
    float mk = (sot <= kDELTA_ART && sood <= kDELTA_NEAR) ? 1.0f : 0.0f;
    mask_g[row0 + m] = mk;
    atomicAdd(gate_g, mk);
  }
  if (tid == 0) { atomicAdd(kl_g, s_kl); atomicAdd(rl_g, s_rl); }
  __syncthreads();

  // ---- Stage G: attention scores, one head per wave ----
  {
    const unsigned short* aWh = aW_s + (size_t)wave * kD * kA;
    v8f av[8] = {};
#pragma unroll
    for (int kt = 0; kt < 8; ++kt) {
      v16bf a = ld_frag(z_s + (kt * 32 + lane) * 16);
#pragma unroll
      for (int t = 0; t < 8; ++t) {
        v16bf b = ld_frag(aWh + ((size_t)(t * 8 + kt) * 32 + lane) * 16);
        av[t] = WMMA_BF16(av[t], a, b);
      }
    }
    float sc[8] = {0, 0, 0, 0, 0, 0, 0, 0};
#pragma unroll
    for (int t = 0; t < 8; ++t) {
      int col = t * 16 + (lane & 15);
      float wb = aWb[wave * kA + col];
      float vv = aV[wave * kA + col];
#pragma unroll
      for (int r = 0; r < 8; ++r) sc[r] += tanhf(av[t][r] + wb) * vv;
    }
#pragma unroll
    for (int r = 0; r < 8; ++r) atomicAdd(&s_score[wave * 16 + mlo + r], sc[r]);
  }
  __syncthreads();
  if (tid < kH * 16) {
    int hh = tid >> 4, m = tid & 15;
    float s = (s_score[tid] + aVb[hh]) / s_ufin[m];     // raw score / u_safe (mask later)
    score_g[(size_t)hh * kN + row0 + m] = s;
  }
}

// ================= K3a: per-head max =================
__global__ void k3_max(const float* __restrict__ score, const float* __restrict__ mask,
                       const float* __restrict__ gcnt, unsigned* __restrict__ hmax) {
  int h = blockIdx.y;
  bool fb = (*gcnt < 1.0f);
  float lm = -3.0e38f;
  for (int n = blockIdx.x * blockDim.x + threadIdx.x; n < kN; n += gridDim.x * blockDim.x) {
    float m = fb ? 1.0f : mask[n];
    lm = fmaxf(lm, score[(size_t)h * kN + n] * m);
  }
  __shared__ float red[256];
  red[threadIdx.x] = lm;
  __syncthreads();
  for (int s = 128; s > 0; s >>= 1) {
    if (threadIdx.x < s) red[threadIdx.x] = fmaxf(red[threadIdx.x], red[threadIdx.x + s]);
    __syncthreads();
  }
  if (threadIdx.x == 0) atomicMax(&hmax[h], ford(red[0]));
}

// ================= K3b: per-head sum(exp) =================
__global__ void k3_sum(const float* __restrict__ score, const float* __restrict__ mask,
                       const float* __restrict__ gcnt, const unsigned* __restrict__ hmax,
                       float* __restrict__ hsum) {
  int h = blockIdx.y;
  bool fb = (*gcnt < 1.0f);
  float mx = funord(hmax[h]);
  float ls = 0.0f;
  for (int n = blockIdx.x * blockDim.x + threadIdx.x; n < kN; n += gridDim.x * blockDim.x) {
    float m = fb ? 1.0f : mask[n];
    ls += __expf(score[(size_t)h * kN + n] * m - mx);
  }
  __shared__ float red[256];
  red[threadIdx.x] = ls;
  __syncthreads();
  for (int s = 128; s > 0; s >>= 1) {
    if (threadIdx.x < s) red[threadIdx.x] += red[threadIdx.x + s];
    __syncthreads();
  }
  if (threadIdx.x == 0) atomicAdd(&hsum[h], red[0]);
}

// ================= K3c: attn, heads = attn @ z, attn_w =================
__global__ void k3_heads(const float* __restrict__ score, const float* __restrict__ mask,
                         const float* __restrict__ gcnt, const unsigned* __restrict__ hmax,
                         const float* __restrict__ hsum, const unsigned short* __restrict__ z_g,
                         float* __restrict__ heads, float* __restrict__ attn_w) {
  int h = blockIdx.y;
  int d = threadIdx.x;                     // 256 cols
  bool fb = (*gcnt < 1.0f);
  float mx = funord(hmax[h]);
  float inv = 1.0f / hsum[h];
  float acc = 0.0f;
  for (int n = blockIdx.x; n < kN; n += gridDim.x) {
    if (n + (int)gridDim.x < kN)
      __builtin_prefetch(&z_g[(size_t)(n + gridDim.x) * kD + d], 0, 1);
    float m = fb ? 1.0f : mask[n];
    float attn = __expf(score[(size_t)h * kN + n] * m - mx) * inv;
    acc += attn * bf2f(z_g[(size_t)n * kD + d]);
    if (d == 0) atomicAdd(&attn_w[n], attn * (1.0f / kH));
  }
  atomicAdd(&heads[h * kD + d], acc);
}

// ================= K4: z_bag, classifier, outputs =================
__global__ void k4_final(const float* __restrict__ heads, const float* __restrict__ hmW,
                         const float* __restrict__ hmb, const float* __restrict__ cng,
                         const float* __restrict__ cnb, const float* __restrict__ cW1,
                         const float* __restrict__ cb1, const float* __restrict__ cW2,
                         const float* __restrict__ cb2, const float* __restrict__ log_temp,
                         const float* __restrict__ klsum, const float* __restrict__ rlsum,
                         float* __restrict__ out) {
  __shared__ float s_ln[kD];
  __shared__ float s_c[kD / 2];
  __shared__ float s_red[2];
  __shared__ float s_logit[2];
  int tid = threadIdx.x;                   // 256 threads

  float zb = hmb[tid];
  for (int j = 0; j < kH * kD; ++j) zb += heads[j] * hmW[(size_t)j * kD + tid];
  out[OUT_ZBAG + tid] = zb;

  if (tid < 2) s_red[tid] = 0.0f;
  __syncthreads();
  atomicAdd(&s_red[0], zb);
  atomicAdd(&s_red[1], zb * zb);
  __syncthreads();
  float mean = s_red[0] * (1.0f / kD);
  float var  = s_red[1] * (1.0f / kD) - mean * mean;
  s_ln[tid] = (zb - mean) * rsqrtf(var + 1e-5f) * cng[tid] + cnb[tid];
  __syncthreads();

  if (tid < kD / 2) {
    float a = cb1[tid];
    for (int dd = 0; dd < kD; ++dd) a += s_ln[dd] * cW1[(size_t)dd * (kD / 2) + tid];
    s_c[tid] = gelu_exact(a);
  }
  __syncthreads();

  if (tid < 2) {
    float l = cb2[tid];
    for (int j = 0; j < kD / 2; ++j) l += s_c[j] * cW2[j * 2 + tid];
    out[OUT_LOGITS + tid] = l;
    s_logit[tid] = l;
  }
  __syncthreads();
  if (tid < 2) {
    float temp = fminf(10.0f, fmaxf(0.01f, __expf(log_temp[0])));
    float a0 = s_logit[0] / temp, a1 = s_logit[1] / temp;
    float mx = fmaxf(a0, a1);
    float e0 = __expf(a0 - mx), e1 = __expf(a1 - mx);
    out[OUT_PHAT + tid] = (tid == 0 ? e0 : e1) / (e0 + e1);
  }
  if (tid == 0) {
    float inv = 1.0f / ((float)kN * (float)kD);
    out[OUT_KL] = -0.5f * (*klsum) * inv;
    out[OUT_RL] = (*rlsum) * inv;
  }
}

// ================= host launcher =================
extern "C" void kernel_launch(void* const* d_in, const int* in_sizes, int n_in,
                              void* d_out, int out_size, void* d_ws, size_t ws_size,
                              hipStream_t stream) {
  (void)in_sizes; (void)n_in; (void)out_size; (void)ws_size;
  const float* feats = (const float*)d_in[0];
  const float* pW  = (const float*)d_in[1];
  const float* pb  = (const float*)d_in[2];
  const float* png = (const float*)d_in[3];
  const float* pnb = (const float*)d_in[4];
  const float* muW = (const float*)d_in[5];
  const float* mub = (const float*)d_in[6];
  const float* lvW = (const float*)d_in[7];
  const float* lvb = (const float*)d_in[8];
  const float* protos = (const float*)d_in[9];
  const float* dW1 = (const float*)d_in[10];
  const float* db1 = (const float*)d_in[11];
  const float* dng = (const float*)d_in[12];
  const float* dnb = (const float*)d_in[13];
  const float* dW2 = (const float*)d_in[14];
  const float* db2 = (const float*)d_in[15];
  const float* aW  = (const float*)d_in[16];
  const float* aWb = (const float*)d_in[17];
  const float* aV  = (const float*)d_in[18];
  const float* aVb = (const float*)d_in[19];
  const float* hmW = (const float*)d_in[20];
  const float* hmb = (const float*)d_in[21];
  const float* cng = (const float*)d_in[22];
  const float* cnb = (const float*)d_in[23];
  const float* cW1 = (const float*)d_in[24];
  const float* cb1 = (const float*)d_in[25];
  const float* cW2 = (const float*)d_in[26];
  const float* cb2 = (const float*)d_in[27];
  const float* log_temp = (const float*)d_in[28];

  float* out = (float*)d_out;
  char* ws = (char*)d_ws;

  // workspace layout (all 16B aligned)
  size_t off = 0;
  auto take = [&](size_t bytes) { size_t o = off; off += (bytes + 15) & ~(size_t)15; return o; };
  const size_t OFF_HSWZ = take((size_t)kN * kD * 2);       // h swizzled bf16
  const size_t OFF_Z    = take((size_t)kN * kD * 2);       // z linear bf16
  const size_t OFF_PNL  = take((size_t)kP * kD * 2);       // pn linear bf16
  const size_t OFF_PNS  = take((size_t)kP * kD * 2);       // pn swizzled
  const size_t OFF_PWS  = take((size_t)kFEAT * kD * 2);
  const size_t OFF_MUS  = take((size_t)kD * kD * 2);
  const size_t OFF_LVS  = take((size_t)kD * kD * 2);
  const size_t OFF_DW1S = take((size_t)kD * kD2 * 2);
  const size_t OFF_DW2S = take((size_t)kD2 * kD * 2);
  const size_t OFF_AWS  = take((size_t)kH * kD * kA * 2);
  const size_t OFF_U    = take((size_t)kN * 4);
  const size_t OFF_MASK = take((size_t)kN * 4);
  const size_t OFF_SCR  = take((size_t)kH * kN * 4);
  const int    ACC_F_CNT = 8 + kH * kD;                    // kl, rl, gate, hsum[4], pad, heads
  const size_t OFF_ACC  = take((size_t)ACC_F_CNT * 4);
  const size_t OFF_HMAX = take((size_t)kH * 4);

  unsigned short* h_swz = (unsigned short*)(ws + OFF_HSWZ);
  unsigned short* z_g   = (unsigned short*)(ws + OFF_Z);
  unsigned short* pn_l  = (unsigned short*)(ws + OFF_PNL);
  unsigned short* pn_s  = (unsigned short*)(ws + OFF_PNS);
  unsigned short* pW_s  = (unsigned short*)(ws + OFF_PWS);
  unsigned short* muW_s = (unsigned short*)(ws + OFF_MUS);
  unsigned short* lvW_s = (unsigned short*)(ws + OFF_LVS);
  unsigned short* dW1_s = (unsigned short*)(ws + OFF_DW1S);
  unsigned short* dW2_s = (unsigned short*)(ws + OFF_DW2S);
  unsigned short* aW_s  = (unsigned short*)(ws + OFF_AWS);
  float* u_g    = (float*)(ws + OFF_U);
  float* mask_g = (float*)(ws + OFF_MASK);
  float* scr_g  = (float*)(ws + OFF_SCR);
  float* accf   = (float*)(ws + OFF_ACC);
  float* kl_g   = accf + 0;
  float* rl_g   = accf + 1;
  float* gate_g = accf + 2;
  float* hsum_g = accf + 3;          // [H]
  float* heads_g = accf + 8;         // [H*D]
  unsigned* hmax_g = (unsigned*)(ws + OFF_HMAX);

  const int rowTiles = kN / 16;      // 6250
  auto swzBlocks = [](int K, int Ncols) { return (((K >> 5) * (Ncols >> 4)) + 3) / 4; };

  k_init<<<(kN + 255) / 256, 256, 0, stream>>>(out + OUT_ATTN, accf, hmax_g, ACC_F_CNT);
  k_pn<<<kP, kD, 0, stream>>>(protos, pn_l);
  k_wswz<<<swzBlocks(kFEAT, kD), 128, 0, stream>>>(pW, pW_s, kFEAT, kD);
  k_wswz<<<swzBlocks(kD, kD), 128, 0, stream>>>(muW, muW_s, kD, kD);
  k_wswz<<<swzBlocks(kD, kD), 128, 0, stream>>>(lvW, lvW_s, kD, kD);
  k_wswz<<<swzBlocks(kD, kD2), 128, 0, stream>>>(dW1, dW1_s, kD, kD2);
  k_wswz<<<swzBlocks(kD2, kD), 128, 0, stream>>>(dW2, dW2_s, kD2, kD);
  for (int h = 0; h < kH; ++h)
    k_wswz<<<swzBlocks(kD, kA), 128, 0, stream>>>(aW + (size_t)h * kD * kA,
                                                  aW_s + (size_t)h * kD * kA, kD, kA);
  k_wswzT<<<swzBlocks(kD, kP), 128, 0, stream>>>(pn_l, pn_s, kD, kP);

  k1_proj<<<rowTiles, 128, 0, stream>>>(feats, pW_s, pb, png, pnb, h_swz);
  k2_main<<<rowTiles, 128, 0, stream>>>(h_swz, muW_s, mub, lvW_s, lvb, pn_s,
                                        dW1_s, db1, dng, dnb, dW2_s, db2,
                                        aW_s, aWb, aV, aVb,
                                        z_g, u_g, mask_g, scr_g, kl_g, rl_g, gate_g);
  k3_max<<<dim3(128, kH), 256, 0, stream>>>(scr_g, mask_g, gate_g, hmax_g);
  k3_sum<<<dim3(128, kH), 256, 0, stream>>>(scr_g, mask_g, gate_g, hmax_g, hsum_g);
  k3_heads<<<dim3(200, kH), kD, 0, stream>>>(scr_g, mask_g, gate_g, hmax_g, hsum_g,
                                             z_g, heads_g, out + OUT_ATTN);
  k4_final<<<1, kD, 0, stream>>>(heads_g, hmW, hmb, cng, cnb, cW1, cb1, cW2, cb2,
                                 log_temp, kl_g, rl_g, out);
}
